// RegionalAttention_32152125178099
// MI455X (gfx1250) — matrix operand order repaired and checked by reading, not compiled, not run on hardware
//
#include <hip/hip_runtime.h>
#include <hip/hip_bf16.h>

typedef __bf16 bf16_t;
typedef __attribute__((ext_vector_type(16))) __bf16 v16bf;
typedef __attribute__((ext_vector_type(8)))  __bf16 v8bf;
typedef __attribute__((ext_vector_type(8)))  float  v8f;

#define RWS 49      // window (region) size
#define CH 512      // channels
#define NH 8        // heads
#define CG 32       // gating bottleneck
#define NWIN 2048   // B * (N/WS)

// ---- WMMA operand loaders (wave32, bf16 16x16x32; contiguous k order, same in A and B) ----

// A-matrix 16x32 (MxK): lanes 0-15 row m0+lp hold K=k0..k0+15,
// lanes 16-31 same rows hold K=k0+16..k0+31.
__device__ __forceinline__ v16bf ldA(const bf16_t* base, int ld, int m0, int k0, int lane) {
    const int half = lane >> 4, lp = lane & 15;
    const bf16_t* p = base + (m0 + lp) * ld + k0 + half * 16;
    return *(const v16bf*)p;              // K = k0+half*16 .. +15    (32B)
}

// B-matrix 32x16 (KxN): column n0+lp; lanes 0-15 K=k0..k0+15, lanes 16-31 K=k0+16..k0+31.
// Source stored "column-of-B = contiguous row" (weight rows / K rows / V^T rows): one 32B read.
__device__ __forceinline__ v16bf ldB(const bf16_t* base, int ld, int n0, int k0, int lane) {
    const int half = lane >> 4, lp = lane & 15;
    return *(const v16bf*)(base + (n0 + lp) * ld + k0 + half * 16);
}

__device__ __forceinline__ v8f wmma_bf(v16bf a, v16bf b, v8f c) {
    return __builtin_amdgcn_wmma_f32_16x16x32_bf16(false, a, false, b, (short)0, c, false, false);
}

// ---- kernel 0: fp32 weights -> bf16 in workspace (L2-resident, 2MB) ----
__global__ void cvt_weights(const float* __restrict__ qkv_w,
                            const float* __restrict__ proj_w,
                            bf16_t* __restrict__ wsb) {
    const int nq = 1536 * CH, np = CH * CH;
    int i = blockIdx.x * blockDim.x + threadIdx.x;
    if (i < nq)            wsb[i] = (bf16_t)qkv_w[i];
    else if (i < nq + np)  wsb[i] = (bf16_t)proj_w[i - nq];
}

// ---- kernel 1: fully fused regional attention, 1 workgroup = 1 window ----
__global__ void __launch_bounds__(256, 1)
regional_attn(const float* __restrict__ x,
              const float* __restrict__ qkv_b,  const float* __restrict__ proj_b,
              const float* __restrict__ w1,     const float* __restrict__ b1,
              const float* __restrict__ bn_g,   const float* __restrict__ bn_b,
              const float* __restrict__ bn_m,   const float* __restrict__ bn_v,
              const float* __restrict__ w2,     const float* __restrict__ b2,
              const bf16_t* __restrict__ qkv_wb,   // 1536 x 512 bf16
              const bf16_t* __restrict__ proj_wb,  // 512 x 512  bf16
              float* __restrict__ out)
{
    extern __shared__ char smem[];
    bf16_t* xg_s = (bf16_t*)(smem);                 // 64x512 bf16  (64KB) gated input, rows 49..63 zero
    bf16_t* o_s  = (bf16_t*)(smem + 65536);         // 64x512 bf16  (64KB) concat head outputs
    bf16_t* q_s  = (bf16_t*)(smem + 131072);        // 2x 64x64 bf16 (16KB) row-major (i,d), pre-scaled
    bf16_t* k_s  = (bf16_t*)(smem + 147456);        // 2x 64x64 bf16 (16KB) row-major (j,d)
    bf16_t* vT_s = (bf16_t*)(smem + 163840);        // 2x 64x64 bf16 (16KB) TRANSPOSED (d,j)
    float*  s_s  = (float*) (smem + 180224);        // 2x 64x64 f32  (32KB) scores
    bf16_t* p_s  = (bf16_t*)(smem + 212992);        // 2x 64x64 bf16 (16KB) softmax probs

    const int tid  = threadIdx.x;
    const int lane = tid & 31;
    const int lp   = lane & 15, half = lane >> 4;
    const int wv   = tid >> 5;          // 8 waves
    const int win  = blockIdx.x;
    const float* xw = x + (size_t)win * RWS * CH;

    // Warm L2 with the streamed bf16 weights (global_prefetch_b8); a few blocks suffice.
    if (blockIdx.x < 16) {
        const char* pw = (const char*)qkv_wb;
        for (int ofs = tid * 128; ofs < (1536 + 512) * CH * 2; ofs += 256 * 128)
            __builtin_prefetch(pw + ofs, 0, 1);
    }

    // ================= Phase 1: spatial gating (fp32 VALU), xg -> LDS bf16 ==============
    // o_s region is free until P*V of head-group 0; reuse as reduction scratch.
    float* red = (float*)o_s;                 // 8 waves * 32*32 f32 = 32KB
    float* hbc = (float*)o_s + 8 * 1024;      // 8 waves * 32 f32

    for (int t = wv; t < RWS; t += 8) {
        float xv[16];
        #pragma unroll
        for (int u = 0; u < 16; ++u) xv[u] = xw[t * CH + lane + 32 * u];

        #pragma unroll 4
        for (int o = 0; o < CG; ++o) {        // partial dot products (lane-sliced channels)
            float acc = 0.f;
            #pragma unroll
            for (int u = 0; u < 16; ++u)
                acc = fmaf(xv[u], w1[o * CH + lane + 32 * u], acc);
            red[wv * 1024 + o * 32 + lane] = acc;
        }
        {   // lane 'o' finalizes hidden[o]: cross-lane sum + BN + exact GELU
            const int o = lane;
            float hsum = b1[o];
            #pragma unroll
            for (int j = 0; j < 32; ++j) hsum += red[wv * 1024 + o * 32 + j];
            hsum = (hsum - bn_m[o]) * rsqrtf(bn_v[o] + 1e-5f) * bn_g[o] + bn_b[o];
            hsum = 0.5f * hsum * (1.0f + erff(hsum * 0.70710678f));
            hbc[wv * 32 + o] = hsum;
        }
        #pragma unroll
        for (int u = 0; u < 16; ++u) {        // y = W2 @ hidden, residual, -> bf16
            const int c = lane + 32 * u;
            float acc = b2[c];
            #pragma unroll
            for (int o = 0; o < CG; ++o)
                acc = fmaf(w2[c * CG + o], hbc[wv * 32 + o], acc);
            xg_s[t * CH + c] = (bf16_t)(xv[u] + acc);
        }
    }
    for (int t = RWS + wv; t < 64; t += 8)          // zero-pad rows 49..63
        for (int c = lane; c < CH; c += 32) xg_s[t * CH + c] = (bf16_t)0.f;
    __syncthreads();

    // ========== Phase 2: attention, TWO heads per group (uniform 8-wave partition) ======
    const float scale = 0.125f;                      // hd^-0.5, hd=64
    for (int g = 0; g < NH / 2; ++g) {
        const int h0 = 2 * g;

        // --- QKV for head pair: 24 N-tiles, wave owns {wv, wv+8, wv+16}.
        //     M-blocked x4 (weights fetched exactly ONCE per WG) and 3 N-tiles in
        //     flight per wave: 12 WMMAs per A-operand fetch.
        {
            int o0j[3], matj[3], hhj[3], ntj[3];
            #pragma unroll
            for (int j = 0; j < 3; ++j) {
                const int idx = wv + 8 * j;           // 0..23
                matj[j] = idx >> 3;                   // 0=Q 1=K 2=V
                hhj[j]  = (idx >> 2) & 1;             // head within pair
                ntj[j]  = idx & 3;                    // 16-col tile within head
                o0j[j]  = matj[j] * 512 + (h0 + hhj[j]) * 64 + ntj[j] * 16;
            }
            v8f acc[3][4] = {};
            for (int k0 = 0; k0 < CH; k0 += 32) {
                v16bf a0 = ldA(xg_s, CH,  0, k0, lane);
                v16bf a1 = ldA(xg_s, CH, 16, k0, lane);
                v16bf a2 = ldA(xg_s, CH, 32, k0, lane);
                v16bf a3 = ldA(xg_s, CH, 48, k0, lane);
                #pragma unroll
                for (int j = 0; j < 3; ++j) {
                    v16bf b = ldB(qkv_wb, CH, o0j[j], k0, lane);
                    acc[j][0] = wmma_bf(a0, b, acc[j][0]);
                    acc[j][1] = wmma_bf(a1, b, acc[j][1]);
                    acc[j][2] = wmma_bf(a2, b, acc[j][2]);
                    acc[j][3] = wmma_bf(a3, b, acc[j][3]);
                }
            }
            #pragma unroll
            for (int j = 0; j < 3; ++j) {
                const float bias = qkv_b[o0j[j] + lp];
                bf16_t* qk = (matj[j] == 0 ? q_s : k_s) + hhj[j] * 4096;
                if (matj[j] == 0) {                   // Q, pre-scaled
                    #pragma unroll
                    for (int mt = 0; mt < 4; ++mt)
                        #pragma unroll
                        for (int r = 0; r < 8; ++r)
                            qk[(mt * 16 + half * 8 + r) * 64 + ntj[j] * 16 + lp] =
                                (bf16_t)((acc[j][mt][r] + bias) * scale);
                } else if (matj[j] == 1) {            // K row-major
                    #pragma unroll
                    for (int mt = 0; mt < 4; ++mt)
                        #pragma unroll
                        for (int r = 0; r < 8; ++r)
                            qk[(mt * 16 + half * 8 + r) * 64 + ntj[j] * 16 + lp] =
                                (bf16_t)(acc[j][mt][r] + bias);
                } else {                              // V transposed: vT[d][j]
                    bf16_t* vt = vT_s + hhj[j] * 4096;
                    #pragma unroll
                    for (int mt = 0; mt < 4; ++mt)
                        #pragma unroll
                        for (int r = 0; r < 8; ++r)
                            vt[(ntj[j] * 16 + lp) * 64 + mt * 16 + half * 8 + r] =
                                (bf16_t)(acc[j][mt][r] + bias);
                }
            }
        }
        __syncthreads();

        // --- S = Q K^T for both heads: 32 tiles, 4 per wave ---
        #pragma unroll
        for (int j = 0; j < 4; ++j) {
            const int idx = wv + 8 * j;               // 0..31
            const int hh = idx >> 4, tt = idx & 15;
            const int mt = tt >> 2, nt = tt & 3;
            const bf16_t* qb = q_s + hh * 4096;
            const bf16_t* kb = k_s + hh * 4096;
            v8f acc = {};
            acc = wmma_bf(ldA(qb, 64, mt * 16,  0, lane), ldB(kb, 64, nt * 16,  0, lane), acc);
            acc = wmma_bf(ldA(qb, 64, mt * 16, 32, lane), ldB(kb, 64, nt * 16, 32, lane), acc);
            float* sb = s_s + hh * 4096;
            #pragma unroll
            for (int r = 0; r < 8; ++r)
                sb[(mt * 16 + half * 8 + r) * 64 + nt * 16 + lp] = acc[r];
        }
        __syncthreads();

        // --- masked softmax over j<49, fp32: 128 rows, 16 per wave ---
        for (int i2 = wv; i2 < 128; i2 += 8) {
            const int hh = i2 >> 6, i = i2 & 63;
            const float* sb = s_s + hh * 4096;
            float f0 = (lane      < RWS) ? sb[i * 64 + lane]      : -1e30f;
            float f1 = (lane + 32 < RWS) ? sb[i * 64 + lane + 32] : -1e30f;
            float mx = fmaxf(f0, f1);
            #pragma unroll
            for (int o = 16; o > 0; o >>= 1) mx = fmaxf(mx, __shfl_xor(mx, o, 32));
            float e0 = (lane      < RWS) ? __expf(f0 - mx) : 0.f;
            float e1 = (lane + 32 < RWS) ? __expf(f1 - mx) : 0.f;
            float sm = e0 + e1;
            #pragma unroll
            for (int o = 16; o > 0; o >>= 1) sm += __shfl_xor(sm, o, 32);
            const float inv = 1.f / sm;
            bf16_t* pb = p_s + hh * 4096;
            pb[i * 64 + lane]      = (bf16_t)(e0 * inv);
            pb[i * 64 + lane + 32] = (bf16_t)(e1 * inv);
        }
        __syncthreads();

        // --- O_h = P V for both heads: 32 tiles, 4 per wave ---
        #pragma unroll
        for (int j = 0; j < 4; ++j) {
            const int idx = wv + 8 * j;
            const int hh = idx >> 4, tt = idx & 15;
            const int mt = tt >> 2, nt = tt & 3;
            const bf16_t* pb = p_s  + hh * 4096;
            const bf16_t* vb = vT_s + hh * 4096;
            v8f acc = {};
            acc = wmma_bf(ldA(pb, 64, mt * 16,  0, lane), ldB(vb, 64, nt * 16,  0, lane), acc);
            acc = wmma_bf(ldA(pb, 64, mt * 16, 32, lane), ldB(vb, 64, nt * 16, 32, lane), acc);
            #pragma unroll
            for (int r = 0; r < 8; ++r)
                o_s[(mt * 16 + half * 8 + r) * CH + (h0 + hh) * 64 + nt * 16 + lp] =
                    (bf16_t)acc[r];
        }
        __syncthreads();                              // q/k/vT/s/p reused next group
    }

    // ===== Phase 3: proj GEMM 64x512x512, M-blocked x4, N-group of 2 tiles ============
    float* outw = out + (size_t)win * RWS * CH;
    #pragma unroll
    for (int gi = 0; gi < 2; ++gi) {
        const int grp = wv + 8 * gi;                  // 0..15, N-cols [grp*32, grp*32+32)
        v8f acc[2][4] = {};
        for (int k0 = 0; k0 < CH; k0 += 32) {
            v16bf a0 = ldA(o_s, CH,  0, k0, lane);
            v16bf a1 = ldA(o_s, CH, 16, k0, lane);
            v16bf a2 = ldA(o_s, CH, 32, k0, lane);
            v16bf a3 = ldA(o_s, CH, 48, k0, lane);
            #pragma unroll
            for (int nb = 0; nb < 2; ++nb) {
                v16bf b = ldB(proj_wb, CH, grp * 32 + nb * 16, k0, lane);
                acc[nb][0] = wmma_bf(a0, b, acc[nb][0]);
                acc[nb][1] = wmma_bf(a1, b, acc[nb][1]);
                acc[nb][2] = wmma_bf(a2, b, acc[nb][2]);
                acc[nb][3] = wmma_bf(a3, b, acc[nb][3]);
            }
        }
        #pragma unroll
        for (int nb = 0; nb < 2; ++nb) {
            const int o0 = grp * 32 + nb * 16;
            const float bias = proj_b[o0 + lp];
            #pragma unroll
            for (int mt = 0; mt < 4; ++mt)
                #pragma unroll
                for (int r = 0; r < 8; ++r) {
                    const int m = mt * 16 + half * 8 + r;
                    if (m < RWS) outw[m * CH + o0 + lp] = acc[nb][mt][r] + bias;
                }
        }
    }
}

extern "C" void kernel_launch(void* const* d_in, const int* in_sizes, int n_in,
                              void* d_out, int out_size, void* d_ws, size_t ws_size,
                              hipStream_t stream) {
    const float* x      = (const float*)d_in[0];
    const float* qkv_w  = (const float*)d_in[1];
    const float* qkv_b  = (const float*)d_in[2];
    const float* proj_w = (const float*)d_in[3];
    const float* proj_b = (const float*)d_in[4];
    const float* w1     = (const float*)d_in[5];
    const float* b1     = (const float*)d_in[6];
    const float* bn_g   = (const float*)d_in[7];
    const float* bn_b   = (const float*)d_in[8];
    const float* bn_m   = (const float*)d_in[9];
    const float* bn_v   = (const float*)d_in[10];
    const float* w2     = (const float*)d_in[11];
    const float* b2     = (const float*)d_in[12];
    float* out = (float*)d_out;

    bf16_t* wsb = (bf16_t*)d_ws;                       // [1536*512] qkv bf16, [512*512] proj bf16
    const int ncvt = 1536 * CH + CH * CH;
    cvt_weights<<<(ncvt + 255) / 256, 256, 0, stream>>>(qkv_w, proj_w, wsb);

    const size_t lds_bytes = 229376;                   // 224KB of the 320KB WGP pool
    regional_attn<<<NWIN, 256, lds_bytes, stream>>>(
        x, qkv_b, proj_b, w1, b1, bn_g, bn_b, bn_m, bn_v, w2, b2,
        wsb, wsb + 1536 * CH, out);
}
